// AutoNER_24343874634294
// MI455X (gfx1250) — matrix operand order repaired and checked by reading, not compiled, not run on hardware
//
#include <hip/hip_runtime.h>
#include <hip/hip_bf16.h>
#include <math.h>

typedef __attribute__((ext_vector_type(16))) _Float16 v16h;
typedef __attribute__((ext_vector_type(8)))  _Float16 v8h;
typedef __attribute__((ext_vector_type(8)))  float    v8f;

constexpr int kBS   = 64;
constexpr int kWSEQ = 256;
constexpr int kCSEQ = 24;
constexpr int kWD   = 200;
constexpr int kCED  = 20;
constexpr int kCH   = 20;
constexpr int kWH   = 50;
constexpr int kHID  = 512;
constexpr int kNW   = kBS * kWSEQ;       // 16384 word positions
constexpr int kMAXRUNS = 128;            // max spans per batch row
constexpr int kMAXS    = kBS * kMAXRUNS; // 8192 static span slots

__device__ inline float sig_(float x) { return 1.0f / (1.0f + __expf(-x)); }

// ---- WMMA A-fragment loader (contiguous k order; B packed the same way) ------------------
// A (16x32 f16, row-major source, leading dim lda): lane m=lane&15 holds row m;
// element e maps to K = k0 + e + 16*(lane>=16).  Per lane this is one
// contiguous 32B run -> two b128 loads (ds_load_b128 when src is LDS).
__device__ inline v16h load_a_frag(const _Float16* a, int lda, int k0, int lane) {
  const int m   = lane & 15;
  const int hi8 = (lane >> 4) << 4;
  const _Float16* p0 = a + m * lda + k0 + hi8;
  const v8h lo = *(const v8h*)p0;
  const v8h hv = *(const v8h*)(p0 + 8);
  return __builtin_shufflevector(lo, hv, 0, 1, 2, 3, 4, 5, 6, 7,
                                 8, 9, 10, 11, 12, 13, 14, 15);
}

// =============================================================================
// Char BiLSTM layer.  Workgroup = 16 sequences, 5 waves (one per 16-col gate
// tile of 4H=80).  Fused GEMM: A=[x_t | h] (16 x 64, LDS), B=[W_ih;W_hh]
// pre-converted into per-wave f16 register fragments.  Runs full 24-step
// recurrence; backward direction handled via index involution.
// =============================================================================
template <int DIN>
__global__ __launch_bounds__(160)
void char_lstm_kernel(const int* __restrict__ cids, const float* __restrict__ char_emb,
                      const _Float16* __restrict__ xin, const int* __restrict__ clen,
                      const float* __restrict__ Wf_ih, const float* __restrict__ Wf_hh,
                      const float* __restrict__ bf,
                      const float* __restrict__ Wb_ih, const float* __restrict__ Wb_hh,
                      const float* __restrict__ bb,
                      _Float16* __restrict__ xout) {
  constexpr int G  = 4 * kCH;  // 80 gate cols
  constexpr int KP = 64;       // padded K (DIN + 20 <= 60)
  __shared__ alignas(16) _Float16 Asm[16 * KP];
  __shared__ float gbuf[16 * G];
  __shared__ float cbuf[16 * kCH];
  __shared__ int   clenS[16];

  const int tid  = threadIdx.x;
  const int lane = tid & 31;
  const int wave = tid >> 5;
  const int dir  = blockIdx.y;
  const int n0   = blockIdx.x * 16;
  const float* W_ih = dir ? Wb_ih : Wf_ih;
  const float* W_hh = dir ? Wb_hh : Wf_hh;
  const float* bias = dir ? bb : bf;

  for (int i = tid; i < 16 * KP; i += 160) Asm[i] = (_Float16)0.f;
  for (int i = tid; i < 16 * kCH; i += 160) cbuf[i] = 0.f;
  if (tid < 16) { int c = clen[n0 + tid]; clenS[tid] = (c > 0) ? c : 1; }
  __syncthreads();

  // Per-wave B fragments (2 K-chunks of 32) for cols [wave*16, wave*16+16).
  const int colBase = wave * 16;
  const int nIdx    = lane & 15;
  const int hi      = lane >> 4;
  const int col     = colBase + nIdx;
  v16h B0, B1;
#pragma unroll
  for (int e = 0; e < 16; ++e) {
    const int k0 = e + 16 * hi;
    const int k1 = 32 + e + 16 * hi;
    float v0 = 0.f, v1 = 0.f;
    if (k0 < DIN) v0 = W_ih[col * DIN + k0];
    else if (k0 < DIN + kCH) v0 = W_hh[col * kCH + (k0 - DIN)];
    if (k1 < DIN) v1 = W_ih[col * DIN + k1];
    else if (k1 < DIN + kCH) v1 = W_hh[col * kCH + (k1 - DIN)];
    B0[e] = (_Float16)v0; B1[e] = (_Float16)v1;
  }
  const float bval = bias[col];

  for (int t = 0; t < kCSEQ; ++t) {
    // Stage x_t (or x_rev(t)) into A cols [0, DIN).
    for (int i = tid; i < 16 * DIN; i += 160) {
      const int r = i / DIN, c = i % DIN;
      const int cl  = clenS[r];
      const int pos = dir ? ((t < cl) ? (cl - 1 - t) : t) : t;
      const int n   = n0 + r;
      float v;
      if (DIN == kCED) {
        int id = cids[n * kCSEQ + pos];
        id = id < 0 ? 0 : (id > 255 ? 255 : id);
        v = char_emb[id * kCED + c];
      } else {
        v = (float)xin[(n * kCSEQ + pos) * (2 * kCH) + c];
      }
      Asm[r * KP + c] = (_Float16)v;
    }
    __syncthreads();

    v8f C;
#pragma unroll
    for (int e = 0; e < 8; ++e) C[e] = bval;
    v16h A0 = load_a_frag(Asm, KP, 0, lane);
    C = __builtin_amdgcn_wmma_f32_16x16x32_f16(false, A0, false, B0, (short)0, C, false, false);
    v16h A1 = load_a_frag(Asm, KP, 32, lane);
    C = __builtin_amdgcn_wmma_f32_16x16x32_f16(false, A1, false, B1, (short)0, C, false, false);

#pragma unroll
    for (int e = 0; e < 8; ++e) {
      const int m = e + 8 * hi;
      float v = C[e];
      v = (col < 2 * kCH || col >= 3 * kCH) ? sig_(v) : tanhf(v);  // i,f,o sig; g tanh
      gbuf[m * G + col] = v;
    }
    __syncthreads();

    for (int i = tid; i < 16 * kCH; i += 160) {
      const int m = i / kCH, j = i % kCH;
      const float ig = gbuf[m * G + j];
      const float fg = gbuf[m * G + kCH + j];
      const float gg = gbuf[m * G + 2 * kCH + j];
      const float og = gbuf[m * G + 3 * kCH + j];
      const float c  = fg * cbuf[m * kCH + j] + ig * gg;
      cbuf[m * kCH + j] = c;
      const float h = og * tanhf(c);
      Asm[m * KP + DIN + j] = (_Float16)h;
      const int cl  = clenS[m];
      const int pos = dir ? ((t < cl) ? (cl - 1 - t) : t) : t;
      xout[((n0 + m) * kCSEQ + pos) * (2 * kCH) + dir * kCH + j] = (_Float16)h;
    }
    __syncthreads();
  }
}

// =============================================================================
// Word BiLSTM layer.  Workgroup = 16 batch rows, 13 waves (4H=200 padded to
// 208 cols).  B fragments (up to 11 K-chunks = 88 VGPRs/wave) held in regs.
// =============================================================================
template <int DIN>
__global__ __launch_bounds__(416)
void word_lstm_kernel(const _Float16* __restrict__ xin, const int* __restrict__ slength,
                      const float* __restrict__ Wf_ih, const float* __restrict__ Wf_hh,
                      const float* __restrict__ bf,
                      const float* __restrict__ Wb_ih, const float* __restrict__ Wb_hh,
                      const float* __restrict__ bb,
                      _Float16* __restrict__ yout) {
  constexpr int H  = kWH;               // 50
  constexpr int G  = 4 * H;             // 200
  constexpr int KP = ((DIN + H + 31) / 32) * 32;  // 352 (L1) / 160 (L2-4)
  constexpr int NK = KP / 32;
  __shared__ alignas(16) _Float16 Asm[16 * KP];
  __shared__ float gbuf[16 * G];
  __shared__ float cbuf[16 * H];
  __shared__ int   slenS[16];

  const int tid  = threadIdx.x;
  const int lane = tid & 31;
  const int wave = tid >> 5;
  const int dir  = blockIdx.y;
  const int r0   = blockIdx.x * 16;
  const float* W_ih = dir ? Wb_ih : Wf_ih;
  const float* W_hh = dir ? Wb_hh : Wf_hh;
  const float* bias = dir ? bb : bf;

  for (int i = tid; i < 16 * KP; i += 416) Asm[i] = (_Float16)0.f;
  for (int i = tid; i < 16 * H; i += 416) cbuf[i] = 0.f;
  if (tid < 16) slenS[tid] = slength[r0 + tid];
  __syncthreads();

  const int colBase = wave * 16;
  const int nIdx    = lane & 15;
  const int hi      = lane >> 4;
  const int col     = colBase + nIdx;
  v16h B[NK];
#pragma unroll
  for (int kc = 0; kc < NK; ++kc) {
#pragma unroll
    for (int e = 0; e < 16; ++e) {
      const int k = kc * 32 + e + 16 * hi;
      float v = 0.f;
      if (col < G) {
        if (k < DIN) v = W_ih[col * DIN + k];
        else if (k < DIN + H) v = W_hh[col * H + (k - DIN)];
      }
      B[kc][e] = (_Float16)v;
    }
  }
  const float bval = (col < G) ? bias[col] : 0.f;

  for (int t = 0; t < kWSEQ; ++t) {
    for (int i = tid; i < 16 * DIN; i += 416) {
      const int r = i / DIN, c = i % DIN;
      const int sl  = slenS[r];
      const int pos = dir ? ((t < sl) ? (sl - 1 - t) : t) : t;
      Asm[r * KP + c] = xin[((size_t)(r0 + r) * kWSEQ + pos) * DIN + c];
    }
    __syncthreads();

    v8f C;
#pragma unroll
    for (int e = 0; e < 8; ++e) C[e] = bval;
#pragma unroll
    for (int kc = 0; kc < NK; ++kc) {
      v16h A = load_a_frag(Asm, KP, kc * 32, lane);
      C = __builtin_amdgcn_wmma_f32_16x16x32_f16(false, A, false, B[kc], (short)0, C, false, false);
    }
#pragma unroll
    for (int e = 0; e < 8; ++e) {
      const int m = e + 8 * hi;
      if (col < G) {
        float v = C[e];
        v = (col < 2 * H || col >= 3 * H) ? sig_(v) : tanhf(v);
        gbuf[m * G + col] = v;
      }
    }
    __syncthreads();

    for (int i = tid; i < 16 * H; i += 416) {
      const int m = i / H, j = i % H;
      const float ig = gbuf[m * G + j];
      const float fg = gbuf[m * G + H + j];
      const float gg = gbuf[m * G + 2 * H + j];
      const float og = gbuf[m * G + 3 * H + j];
      const float c  = fg * cbuf[m * H + j] + ig * gg;
      cbuf[m * H + j] = c;
      const float h = og * tanhf(c);
      Asm[m * KP + DIN + j] = (_Float16)h;
      const int sl  = slenS[m];
      const int pos = dir ? ((t < sl) ? (sl - 1 - t) : t) : t;
      yout[((size_t)(r0 + m) * kWSEQ + pos) * 100 + dir * H + j] = (_Float16)h;
    }
    __syncthreads();
  }
}

// ---- word input assembly: [word_emb(200) | char_first(40) | char_last(40)] --
__global__ void build_word_input_kernel(const int* __restrict__ ids,
                                        const float* __restrict__ word_emb,
                                        const _Float16* __restrict__ cout2,
                                        const int* __restrict__ clen,
                                        _Float16* __restrict__ xw) {
  const int n = blockIdx.x;
  const int tid = threadIdx.x;
  int id = ids[n]; if (id == -100) id = 0;
  const int cl = clen[n];
  const bool ok = cl > 0;
  const int cl1 = ok ? cl : 1;
  _Float16* dst = xw + (size_t)n * 280;
  for (int c = tid; c < kWD; c += 128) dst[c] = (_Float16)word_emb[(size_t)id * kWD + c];
  for (int c = tid; c < 2 * kCH; c += 128) {
    const float f = ok ? (float)cout2[(n * kCSEQ + 0) * (2 * kCH) + c] : 0.f;
    const float l = ok ? (float)cout2[(n * kCSEQ + (cl1 - 1)) * (2 * kCH) + c] : 0.f;
    dst[kWD + c]           = (_Float16)f;
    dst[kWD + 2 * kCH + c] = (_Float16)l;
  }
}

// ---- mask final BiLSTM output; produce f32 copy + K-padded f16 copy ---------
__global__ void mask_out_kernel(const _Float16* __restrict__ y4, const int* __restrict__ slength,
                                float* __restrict__ outf, _Float16* __restrict__ outh) {
  const int idx = blockIdx.x * blockDim.x + threadIdx.x;
  if (idx >= kNW * 128) return;
  const int n = idx >> 7, c = idx & 127;
  const int b = n / kWSEQ, t = n % kWSEQ;
  float v = 0.f;
  if (c < 100 && t < slength[b]) v = (float)y4[(size_t)n * 100 + c];
  if (c < 100) outf[(size_t)n * 100 + c] = v;
  outh[(size_t)idx] = (_Float16)v;
}

// ---- pack span_w1 into per-lane B-fragment order --------------------------
// dst index = (((jt*4 + kc)*32 + lane)*16 + e);  value = w1[col][k] with
// col = jt*16 + (lane&15), k = kc*32 + e + 16*(lane>=16)  (zero-padded k>=100).
__global__ void prep_w1_kernel(const float* __restrict__ w1, _Float16* __restrict__ w1h) {
  const int idx = blockIdx.x * blockDim.x + threadIdx.x;
  if (idx >= 32 * 4 * 32 * 16) return;
  const int e    = idx & 15;
  const int lane = (idx >> 4) & 31;
  const int kc   = (idx >> 9) & 3;
  const int jt   = idx >> 11;
  const int col  = jt * 16 + (lane & 15);
  const int k    = kc * 32 + e + 16 * (lane >> 4);
  w1h[idx] = (_Float16)((k < 100) ? w1[col * 100 + k] : 0.f);
}

// ---- span hidden layer: relu(out(16384x128) @ w1.T + b1), WMMA --------------
__global__ __launch_bounds__(128)
void span_gemm_kernel(const _Float16* __restrict__ outh, const _Float16* __restrict__ w1h,
                      const float* __restrict__ b1, _Float16* __restrict__ hsp) {
  const int lane = threadIdx.x & 31, wave = threadIdx.x >> 5;
  const int m0 = blockIdx.x * 16;
  const int jt = blockIdx.y * 4 + wave;   // 16-col tile index (0..31)
  const int j0 = jt * 16;
  const int nIdx = lane & 15, hi = lane >> 4;
  const int col = j0 + nIdx;
  v8f C;
  const float bv = b1[col];
#pragma unroll
  for (int e = 0; e < 8; ++e) C[e] = bv;
#pragma unroll
  for (int kc = 0; kc < 4; ++kc) {
    v16h A = load_a_frag(outh + (size_t)m0 * 128, 128, kc * 32, lane);
    // fragment-ordered pack: one contiguous 32B load per lane
    v16h Bf = *(const v16h*)(w1h + ((size_t)((jt * 4 + kc) * 32 + lane) << 4));
    C = __builtin_amdgcn_wmma_f32_16x16x32_f16(false, A, false, Bf, (short)0, C, false, false);
  }
#pragma unroll
  for (int e = 0; e < 8; ++e) {
    const int m = e + 8 * hi;
    float v = C[e];
    hsp[(size_t)(m0 + m) * kHID + col] = (_Float16)(v > 0.f ? v : 0.f);
  }
}

// ---- span logits (N=2) + per-row NLL (deterministic, no atomics) ------------
__global__ __launch_bounds__(256)
void span_logits_kernel(const _Float16* __restrict__ hsp, const float* __restrict__ w2,
                        const float* __restrict__ b2, const int* __restrict__ span_label,
                        float* __restrict__ nllArr) {
  const int wave = threadIdx.x >> 5, lane = threadIdx.x & 31;
  const int row = blockIdx.x * 8 + wave;
  const _Float16* h = hsp + (size_t)row * kHID;
  float s0 = 0.f, s1 = 0.f;
  for (int k = lane; k < kHID; k += 32) {
    const float hv = (float)h[k];
    s0 += hv * w2[k];
    s1 += hv * w2[kHID + k];
  }
#pragma unroll
  for (int off = 16; off; off >>= 1) {
    s0 += __shfl_xor(s0, off, 32);
    s1 += __shfl_xor(s1, off, 32);
  }
  if (lane == 0) {
    s0 += b2[0]; s1 += b2[1];
    const int lab = span_label[row];
    float r = 0.f;
    if (lab != -100) {
      const float m = fmaxf(s0, s1);
      const float lse = m + logf(__expf(s0 - m) + __expf(s1 - m));
      r = lse - (lab ? s1 : s0);
    }
    nllArr[row] = r;
  }
}

// ---- span enumeration (matches _spans_host); static slots -> deterministic --
__global__ void span_enum_kernel(const int* __restrict__ span_label, const int* __restrict__ bio0,
                                 const int* __restrict__ bio1, const int* __restrict__ bio2,
                                 int* __restrict__ sB, int* __restrict__ sS, int* __restrict__ sE,
                                 int* __restrict__ sV, float* __restrict__ sT) {
  const int b = threadIdx.x;
  if (b >= kBS) return;
  int cnt = 0, rs = -1;
  bool a0 = true, a1 = true, a2 = true;
  for (int t = 0; t <= kWSEQ; ++t) {
    const bool one = (t < kWSEQ) && (span_label[b * kWSEQ + t] == 1);
    if (one) {
      if (rs < 0) { rs = t; a0 = a1 = a2 = true; }
      a0 = a0 && (bio0[b * kWSEQ + t] == 0);
      a1 = a1 && (bio1[b * kWSEQ + t] == 1);
      a2 = a2 && (bio2[b * kWSEQ + t] == 2);
    } else if (rs >= 0) {
      if (cnt < kMAXRUNS) {
        const int slot = b * kMAXRUNS + cnt;
        const bool k2 = a2 && !(a0 || a1);   // remove label 2 if 0/1 present
        float t0 = a0 ? 1.f : 0.f, t1 = a1 ? 1.f : 0.f, t2 = k2 ? 1.f : 0.f;
        const float nrm = sqrtf(t0 + t1 + t2);
        if (nrm > 0.f) { t0 /= nrm; t1 /= nrm; t2 /= nrm; }
        sB[slot] = b; sS[slot] = rs; sE[slot] = t; sV[slot] = 1;
        sT[slot * 3 + 0] = t0; sT[slot * 3 + 1] = t1; sT[slot * 3 + 2] = t2;
        ++cnt;
      }
      rs = -1;
    }
  }
  for (int i = cnt; i < kMAXRUNS; ++i) sV[b * kMAXRUNS + i] = 0;
}

// ---- entity head: feat=[head|mean|tail](300) -> relu(512) -> 3 logits -------
__global__ __launch_bounds__(128)
void ent_mlp_kernel(const int* __restrict__ sB, const int* __restrict__ sS,
                    const int* __restrict__ sE, const int* __restrict__ sV,
                    const float* __restrict__ sT, const float* __restrict__ outf,
                    const float* __restrict__ w1, const float* __restrict__ b1,
                    const float* __restrict__ w2, const float* __restrict__ b2,
                    float* __restrict__ lossArr) {
  const int slot = blockIdx.x;
  const int tid  = threadIdx.x;
  __shared__ float feat[304];
  __shared__ float hbuf[kHID];
  __shared__ float ered[4];
  if (!sV[slot]) { if (tid == 0) lossArr[slot] = 0.f; return; }
  const int b = sB[slot], s = sS[slot], e = sE[slot];
  const float* base = outf + (size_t)b * kWSEQ * 100;
  for (int c = tid; c < 100; c += 128) {
    float acc = 0.f;
    for (int t = s; t < e; ++t) acc += base[t * 100 + c];
    feat[c]       = base[s * 100 + c];
    feat[100 + c] = acc / (float)(e - s);
    feat[200 + c] = base[(e - 1) * 100 + c];
  }
  __syncthreads();
  for (int j = tid; j < kHID; j += 128) {
    float acc = b1[j];
    const float* wr = w1 + (size_t)j * 300;
    for (int k = 0; k < 300; ++k) acc += feat[k] * wr[k];
    hbuf[j] = acc > 0.f ? acc : 0.f;
  }
  __syncthreads();
  const int wave = tid >> 5, lane = tid & 31;
  if (wave < 3) {
    float acc = 0.f;
    const float* wr = w2 + (size_t)wave * kHID;
    for (int k = lane; k < kHID; k += 32) acc += hbuf[k] * wr[k];
#pragma unroll
    for (int off = 16; off; off >>= 1) acc += __shfl_xor(acc, off, 32);
    if (lane == 0) ered[wave] = acc + b2[wave];
  }
  __syncthreads();
  if (tid == 0) {
    const float e0 = ered[0], e1 = ered[1], e2 = ered[2];
    const float m = fmaxf(e0, fmaxf(e1, e2));
    const float lse = m + logf(__expf(e0 - m) + __expf(e1 - m) + __expf(e2 - m));
    lossArr[slot] = -(sT[slot * 3 + 0] * (e0 - lse) + sT[slot * 3 + 1] * (e1 - lse) +
                      sT[slot * 3 + 2] * (e2 - lse));
  }
}

// ---- fixed-order reductions -> d_out = [span_loss, ent_loss] ----------------
__global__ __launch_bounds__(256)
void finalize_kernel(const float* __restrict__ nllArr, const int* __restrict__ span_label,
                     const float* __restrict__ lossArr, float* __restrict__ out) {
  __shared__ float p[256], q[256], r[256];
  const int tid = threadIdx.x;
  float sn = 0.f, sc = 0.f, se = 0.f;
  for (int i = tid; i < kNW; i += 256) {
    sn += nllArr[i];
    sc += (span_label[i] != -100) ? 1.f : 0.f;
  }
  for (int i = tid; i < kMAXS; i += 256) se += lossArr[i];
  p[tid] = sn; q[tid] = sc; r[tid] = se;
  __syncthreads();
  if (tid == 0) {
    float a = 0.f, bsum = 0.f, c = 0.f;
    for (int i = 0; i < 256; ++i) { a += p[i]; bsum += q[i]; c += r[i]; }
    out[0] = a / bsum;
    out[1] = c;
  }
}

// =============================================================================
extern "C" void kernel_launch(void* const* d_in, const int* in_sizes, int n_in,
                              void* d_out, int out_size, void* d_ws, size_t ws_size,
                              hipStream_t stream) {
  (void)in_sizes; (void)n_in; (void)out_size; (void)ws_size;
  const int* input_ids  = (const int*)d_in[0];
  const int* char_ids   = (const int*)d_in[1];
  const int* char_len   = (const int*)d_in[2];
  const int* slength    = (const int*)d_in[3];
  const int* span_label = (const int*)d_in[4];
  const int* bio0       = (const int*)d_in[5];
  const int* bio1       = (const int*)d_in[6];
  const int* bio2       = (const int*)d_in[7];
  int w = 8;
  const float* word_emb = (const float*)d_in[w++];
  const float* char_emb = (const float*)d_in[w++];
  const float* cl_w[2][6];   // {Wf_ih, Wf_hh, bf, Wb_ih, Wb_hh, bb}
  for (int L = 0; L < 2; ++L) for (int k = 0; k < 6; ++k) cl_w[L][k] = (const float*)d_in[w++];
  const float* wl_w[4][6];
  for (int L = 0; L < 4; ++L) for (int k = 0; k < 6; ++k) wl_w[L][k] = (const float*)d_in[w++];
  const float* span_w1 = (const float*)d_in[w++];
  const float* span_b1 = (const float*)d_in[w++];
  const float* span_w2 = (const float*)d_in[w++];
  const float* span_b2 = (const float*)d_in[w++];
  const float* ent_w1  = (const float*)d_in[w++];
  const float* ent_b1  = (const float*)d_in[w++];
  const float* ent_w2  = (const float*)d_in[w++];
  const float* ent_b2  = (const float*)d_in[w++];

  // workspace bump allocator
  char* p = (char*)d_ws;
  auto alloc = [&](size_t bytes) -> void* {
    void* r = (void*)p;
    p += (bytes + 255) & ~(size_t)255;
    return r;
  };
  _Float16* x1   = (_Float16*)alloc((size_t)kNW * kCSEQ * 40 * 2);
  _Float16* x2   = (_Float16*)alloc((size_t)kNW * kCSEQ * 40 * 2);
  _Float16* xw   = (_Float16*)alloc((size_t)kNW * 280 * 2);
  _Float16* yA   = (_Float16*)alloc((size_t)kNW * 100 * 2);
  _Float16* yB   = (_Float16*)alloc((size_t)kNW * 100 * 2);
  float*    outf = (float*)   alloc((size_t)kNW * 100 * 4);
  _Float16* outh = (_Float16*)alloc((size_t)kNW * 128 * 2);
  _Float16* w1h  = (_Float16*)alloc((size_t)128 * kHID * 2);
  _Float16* hsp  = (_Float16*)alloc((size_t)kNW * kHID * 2);
  float* nllArr  = (float*)alloc((size_t)kNW * 4);
  float* lossArr = (float*)alloc((size_t)kMAXS * 4);
  int* sB = (int*)alloc((size_t)kMAXS * 4);
  int* sS = (int*)alloc((size_t)kMAXS * 4);
  int* sE = (int*)alloc((size_t)kMAXS * 4);
  int* sV = (int*)alloc((size_t)kMAXS * 4);
  float* sT = (float*)alloc((size_t)kMAXS * 3 * 4);

  // char BiLSTM (2 layers), each: 1024 row-tiles x 2 directions, 5 waves/wg
  char_lstm_kernel<kCED><<<dim3(kNW / 16, 2), 160, 0, stream>>>(
      char_ids, char_emb, (const _Float16*)nullptr, char_len,
      cl_w[0][0], cl_w[0][1], cl_w[0][2], cl_w[0][3], cl_w[0][4], cl_w[0][5], x1);
  char_lstm_kernel<40><<<dim3(kNW / 16, 2), 160, 0, stream>>>(
      char_ids, char_emb, x1, char_len,
      cl_w[1][0], cl_w[1][1], cl_w[1][2], cl_w[1][3], cl_w[1][4], cl_w[1][5], x2);

  build_word_input_kernel<<<kNW, 128, 0, stream>>>(input_ids, word_emb, x2, char_len, xw);

  // word BiLSTM (4 layers), each: 4 row-tiles x 2 directions, 13 waves/wg
  word_lstm_kernel<280><<<dim3(4, 2), 416, 0, stream>>>(
      xw, slength, wl_w[0][0], wl_w[0][1], wl_w[0][2], wl_w[0][3], wl_w[0][4], wl_w[0][5], yA);
  word_lstm_kernel<100><<<dim3(4, 2), 416, 0, stream>>>(
      yA, slength, wl_w[1][0], wl_w[1][1], wl_w[1][2], wl_w[1][3], wl_w[1][4], wl_w[1][5], yB);
  word_lstm_kernel<100><<<dim3(4, 2), 416, 0, stream>>>(
      yB, slength, wl_w[2][0], wl_w[2][1], wl_w[2][2], wl_w[2][3], wl_w[2][4], wl_w[2][5], yA);
  word_lstm_kernel<100><<<dim3(4, 2), 416, 0, stream>>>(
      yA, slength, wl_w[3][0], wl_w[3][1], wl_w[3][2], wl_w[3][3], wl_w[3][4], wl_w[3][5], yB);

  mask_out_kernel<<<(kNW * 128 + 255) / 256, 256, 0, stream>>>(yB, slength, outf, outh);
  prep_w1_kernel<<<(32 * 4 * 32 * 16 + 255) / 256, 256, 0, stream>>>(span_w1, w1h);
  span_gemm_kernel<<<dim3(kNW / 16, kHID / 64), 128, 0, stream>>>(outh, w1h, span_b1, hsp);
  span_logits_kernel<<<kNW / 8, 256, 0, stream>>>(hsp, span_w2, span_b2, span_label, nllArr);
  span_enum_kernel<<<1, 64, 0, stream>>>(span_label, bio0, bio1, bio2, sB, sS, sE, sV, sT);
  ent_mlp_kernel<<<kMAXS, 128, 0, stream>>>(sB, sS, sE, sV, sT, outf,
                                            ent_w1, ent_b1, ent_w2, ent_b2, lossArr);
  finalize_kernel<<<1, 256, 0, stream>>>(nllArr, span_label, lossArr, (float*)d_out);
}
